// MultiViewMultiPersonPoseNet_73856257622586
// MI455X (gfx1250) — matrix operand order repaired and checked by reading, not compiled, not run on hardware
//
#include <hip/hip_runtime.h>
#include <hip/hip_bf16.h>
#include <math.h>

#define V_   5
#define B_   32
#define NP_  20
#define NJ_  17
#define NB_  16
#define ND_  128
#define HID  1024
#define HR   1024
#define KIN  33      // NJ + NB input channels to conv head
#define KTOT 99      // KIN*3 (im2col K)
#define KPAD 128     // padded K for 4 x K=32 WMMA steps
#define AST  136     // LDS act-row stride in halves (bank-conflict pad)

typedef _Float16 v8h  __attribute__((ext_vector_type(8)));
typedef _Float16 v16h __attribute__((ext_vector_type(16)));
typedef float    v8f  __attribute__((ext_vector_type(8)));
typedef float    v4f  __attribute__((ext_vector_type(4)));

// BONES is a module-level constant in the reference; bake it in so bone loops
// index register arrays with compile-time constants (no scratch spills).
constexpr int BA[NB_] = {0,0,1,2,5,5,7,6,8, 5, 6,11,11,13,12,14};
constexpr int BB[NB_] = {1,2,3,4,6,7,9,8,10,11,12,12,13,15,14,16};

// ---------------------------------------------------------------------------
// Kernel 0: repack Wc1 (1024 x 33 x 3 f32) into f16 WMMA *A*-fragments
// (transposed GEMM: M = hidden channel n, K = im2col k).
// ISA 16-bit A layout: lane -> M = lane%16;
//   element e -> K = (e<8 ? 0 : 16) + (lane<16 ? 0 : 8) + (e&7)
// Stored per-lane contiguous: afrag[kt(4)][ntile(64)][lane(32)][e(16)],
// so each lane loads one 32B chunk per fragment.
// ---------------------------------------------------------------------------
__global__ __launch_bounds__(256) void prep_afrag(const float* __restrict__ Wc1,
                                                  _Float16* __restrict__ afrag) {
  int gid = blockIdx.x * 256 + threadIdx.x;          // 4*64*32*16 = 131072
  if (gid >= 4 * 64 * 32 * 16) return;
  int e     = gid & 15;
  int lane  = (gid >> 4) & 31;
  int ntile = (gid >> 9) & 63;
  int kt    = gid >> 15;
  int n = ntile * 16 + (lane & 15);
  int k = kt * 32 + ((e < 8) ? 0 : 16) + ((lane < 16) ? 0 : 8) + (e & 7);
  float v = (k < KTOT) ? Wc1[n * KTOT + k] : 0.f;
  afrag[gid] = (_Float16)v;
}

// ---------------------------------------------------------------------------
// Kernel 1: rough-depth MLP.  One block per (b,p) row.
// h = relu(x @ Wr1^T + br1); rough = (h @ Wr2^T + br2) * 1000
// ---------------------------------------------------------------------------
__global__ __launch_bounds__(256) void rough_kernel(
    const float* __restrict__ kpts, const float* __restrict__ imwh,
    const float* __restrict__ Wr1, const float* __restrict__ br1,
    const float* __restrict__ Wr2, const float* __restrict__ br2,
    float* __restrict__ rough_ws) {
  int m = blockIdx.x;                                // 0..639
  int b = m / NP_, p = m % NP_;
  int t = threadIdx.x;
  __shared__ float xs[NJ_ * 2];
  __shared__ float hs[HR];
  __shared__ float part[NJ_][8];

  if (t < NJ_ * 2) {
    float wh = imwh[t & 1];                          // image_wh[0,0,:]
    xs[t] = kpts[(((size_t)b * NP_ + p) * NJ_) * 2 + t] / wh;
  }
  __syncthreads();

  for (int n = t; n < HR; n += 256) {
    float s = br1[n];
    const float* wrow = Wr1 + n * (NJ_ * 2);
#pragma unroll
    for (int k = 0; k < NJ_ * 2; ++k) s += xs[k] * wrow[k];
    hs[n] = fmaxf(s, 0.f);
  }
  __syncthreads();

  if (t < NJ_ * 8) {
    int j = t >> 3, r = t & 7;
    float s = 0.f;
    const float* w2 = Wr2 + j * HR + r * 128;
    const float* hp = hs + r * 128;
    for (int k = 0; k < 128; ++k) s += hp[k] * w2[k];
    part[j][r] = s;
  }
  __syncthreads();

  if (t < NJ_) {
    float s = br2[t];
#pragma unroll
    for (int r = 0; r < 8; ++r) s += part[t][r];
    rough_ws[m * NJ_ + t] = s * 1000.f;
  }
}

// ---------------------------------------------------------------------------
// Kernel 2: multi-view scoring.  One block per (b,p); one thread per depth bin.
// Produces xin (640, 33, 128) = [psv | psv2].
// ---------------------------------------------------------------------------
__global__ __launch_bounds__(128) void score_kernel(
    const float* __restrict__ kpts, const float* __restrict__ jvis,
    const float* __restrict__ rough_ws,
    const float* __restrict__ camR, const float* __restrict__ camT,
    const float* __restrict__ camF, const float* __restrict__ camC,
    const float* __restrict__ imwh, const int* __restrict__ nper,
    float* __restrict__ xin) {
  int m = blockIdx.x;
  int b = m / NP_, p = m % NP_;
  int t = threadIdx.x;                               // depth bin 0..127

  __shared__ float kt_s[NJ_ * 2];
  __shared__ float rough_s[NJ_];
  __shared__ float pr_s[NP_ * NJ_ * 2];
  __shared__ float vr_s[NP_ * NJ_];

  if (t < NJ_ * 2) kt_s[t] = kpts[(((size_t)b * NP_ + p) * NJ_) * 2 + t];
  if (t < NJ_)     rough_s[t] = rough_ws[m * NJ_ + t];
  __syncthreads();

  // view-0 camera (uniform across block)
  const float* R0 = camR + (size_t)b * 9;
  float r0[9];
#pragma unroll
  for (int i = 0; i < 9; ++i) r0[i] = R0[i];
  float T0x = camT[(size_t)b * 3 + 0], T0y = camT[(size_t)b * 3 + 1], T0z = camT[(size_t)b * 3 + 2];
  float f0x = camF[(size_t)b * 2 + 0], f0y = camF[(size_t)b * 2 + 1];
  float c0x = camC[(size_t)b * 2 + 0], c0y = camC[(size_t)b * 2 + 1];

  float label = 2000.f + (float)t * (6000.f / 127.f);

  float p3x[NJ_], p3y[NJ_], p3z[NJ_];
#pragma unroll
  for (int j = 0; j < NJ_; ++j) {
    float dep = label + rough_s[j];
    float ux = (kt_s[2 * j]     - c0x) / f0x;
    float uy = (kt_s[2 * j + 1] - c0y) / f0y;
    float cx = ux * dep, cy = uy * dep, cz = dep;
    // p3d = R0^T * cam + T0   (einsum 'bji,bpkdj->bpkdi')
    p3x[j] = r0[0] * cx + r0[3] * cy + r0[6] * cz + T0x;
    p3y[j] = r0[1] * cx + r0[4] * cy + r0[7] * cz + T0y;
    p3z[j] = r0[2] * cx + r0[5] * cy + r0[8] * cz + T0z;
  }

  float accS[NJ_], accB[NJ_], accS2[NB_], accB2[NB_];
#pragma unroll
  for (int j = 0; j < NJ_; ++j) { accS[j] = 0.f; accB[j] = 0.f; }
#pragma unroll
  for (int nb = 0; nb < NB_; ++nb) { accS2[nb] = 0.f; accB2[nb] = 0.f; }

  for (int rv = 1; rv < V_; ++rv) {
    __syncthreads();                                  // protect pr_s/vr_s reuse
    for (int i = t; i < NP_ * NJ_ * 2; i += 128)
      pr_s[i] = kpts[((size_t)rv * B_ + b) * NP_ * NJ_ * 2 + i];
    for (int i = t; i < NP_ * NJ_; i += 128)
      vr_s[i] = jvis[((size_t)rv * B_ + b) * NP_ * NJ_ + i];
    __syncthreads();

    const float* Rr = camR + ((size_t)rv * B_ + b) * 9;
    float rr[9];
#pragma unroll
    for (int i = 0; i < 9; ++i) rr[i] = Rr[i];
    float Trx = camT[((size_t)rv * B_ + b) * 3 + 0];
    float Try = camT[((size_t)rv * B_ + b) * 3 + 1];
    float Trz = camT[((size_t)rv * B_ + b) * 3 + 2];
    float frx = camF[((size_t)rv * B_ + b) * 2 + 0];
    float fry = camF[((size_t)rv * B_ + b) * 2 + 1];
    float crx = camC[((size_t)rv * B_ + b) * 2 + 0];
    float cry = camC[((size_t)rv * B_ + b) * 2 + 1];
    float wim = imwh[((size_t)rv * B_ + b) * 2 + 0];
    float him = imwh[((size_t)rv * B_ + b) * 2 + 1];
    int npv = nper[rv * B_ + b];

    float p2x[NJ_], p2y[NJ_];
#pragma unroll
    for (int j = 0; j < NJ_; ++j) {
      float dx = p3x[j] - Trx, dy = p3y[j] - Try, dz = p3z[j] - Trz;
      float xcx = rr[0] * dx + rr[1] * dy + rr[2] * dz;
      float xcy = rr[3] * dx + rr[4] * dy + rr[5] * dz;
      float xcz = rr[6] * dx + rr[7] * dy + rr[8] * dz;
      float iz = 1.f / (xcz + 1e-8f);
      p2x[j] = xcx * iz * frx + crx;
      p2y[j] = xcy * iz * fry + cry;
    }

    // person matching: argmin over q of visibility-weighted mean sq dist
    float best = 3.4e38f;
    int mm = 0;
    for (int q = 0; q < NP_; ++q) {
      float s = 0.f, ws = 0.f;
#pragma unroll
      for (int j = 0; j < NJ_; ++j) {
        float w = vr_s[q * NJ_ + j];
        float ddx = p2x[j] - pr_s[(q * NJ_ + j) * 2];
        float ddy = p2y[j] - pr_s[(q * NJ_ + j) * 2 + 1];
        s += w * (ddx * ddx + ddy * ddy);
        ws += w;
      }
      float pd = s / (ws + 1e-8f);
      if (q >= npv) pd = 100000.f;
      if (pd < best) { best = pd; mm = q; }
    }

    const float* mpr = &pr_s[mm * NJ_ * 2];
    const float* mvr = &vr_s[mm * NJ_];
#pragma unroll
    for (int j = 0; j < NJ_; ++j) {
      float ddx = p2x[j] - mpr[2 * j];
      float ddy = p2y[j] - mpr[2 * j + 1];
      float md = ddx * ddx + ddy * ddy;
      float sc = __expf(-sqrtf(md + 1e-12f) * (1.f / 100.f));
      float inb = (p2x[j] >= 0.f && p2y[j] >= 0.f &&
                   p2x[j] <= wim - 1.f && p2y[j] <= him - 1.f) ? 1.f : 0.f;
      float bd = inb * mvr[j];
      accS[j] += sc * bd;
      accB[j] += bd;
    }
    float b2 = mvr[0];                               // mvis[:, :, :1, :] broadcast
#pragma unroll
    for (int nb = 0; nb < NB_; ++nb) {
      float mx = mpr[2 * BA[nb]] - mpr[2 * BB[nb]];
      float my = mpr[2 * BA[nb] + 1] - mpr[2 * BB[nb] + 1];
      float blm = sqrtf(mx * mx + my * my + 1e-12f);
      float px = p2x[BA[nb]] - p2x[BB[nb]];
      float py = p2y[BA[nb]] - p2y[BB[nb]];
      float blp = sqrtf(px * px + py * py + 1e-12f);
      float sc2 = __expf(-fabsf(blm - blp) * (1.f / 5.f));
      accS2[nb] += sc2 * b2;
      accB2[nb] += b2;
    }
  }

  float* xrow = xin + (size_t)m * KIN * ND_;
#pragma unroll
  for (int j = 0; j < NJ_; ++j)
    xrow[j * ND_ + t] = accS[j] / (accB[j] + 1e-8f);
#pragma unroll
  for (int nb = 0; nb < NB_; ++nb)
    xrow[(NJ_ + nb) * ND_ + t] = accS2[nb] / (accB2[nb] + 1e-8f);
}

// ---------------------------------------------------------------------------
// Kernel 3: WMMA conv head (transposed GEMM) + fused conv2 + softmax head.
// D[M=n, N=d] = W[n,k] x im2col[k,d]:  A = weights (regs), B = acts (LDS).
// C layout: lane -> one fixed d; VGPR r -> n = ntile*16 + r + (lane<16?0:8).
// conv2 reduces over n IN REGISTERS (24 FMA), then 3 conflict-free
// ds_add_f32 per lane into half-wave-private accumulators.
// ---------------------------------------------------------------------------
__global__ __launch_bounds__(256) void conv_head(
    const float* __restrict__ xin, const _Float16* __restrict__ afrag,
    const float* __restrict__ bc1, const float* __restrict__ Wc2,
    const float* __restrict__ bc2, float* __restrict__ out) {
  __shared__ __align__(16) _Float16 act_lds[ND_ * AST];  // 34,816 B im2col
  __shared__ __align__(16) float bc1_s[HID];
  __shared__ __align__(16) float w2_s[HID * 3];
  __shared__ float logit_s[2][ND_ + 4];                  // per-half-wave accum
  __shared__ float red_v[128];
  __shared__ int   red_i[128];

  int m = blockIdx.x;
  int t = threadIdx.x;
  const float* xrow = xin + (size_t)m * KIN * ND_;

  // Build im2col (f32 -> f16), zero-padded in K (99->128) and d edges.
  for (int idx = t; idx < ND_ * KPAD; idx += 256) {
    int d = idx >> 7, k = idx & 127;
    float v = 0.f;
    if (k < KTOT) {
      int c = k / 3, to = k - 3 * c;
      int din = d + to - 1;                            // SAME pad, kernel=3
      if (din >= 0 && din < ND_) v = xrow[c * ND_ + din];
    }
    act_lds[d * AST + k] = (_Float16)v;
  }
  for (int i = t; i < HID; i += 256) bc1_s[i] = bc1[i];
  for (int i = t; i < HID * 3; i += 256) w2_s[i] = Wc2[i];
  if (t < ND_ + 4) { logit_s[0][t] = 0.f; logit_s[1][t] = 0.f; }
  __syncthreads();

  int wave = t >> 5, lane = t & 31;
  int nlocal = lane & 15;
  int half  = (lane < 16) ? 0 : 1;
  int hb8   = half * 8;                               // M (n) half offset
  int koffB = half * 8;                               // B-frag K half offset

  for (int nti = 0; nti < 8; ++nti) {
    int ntile = wave * 8 + nti;
    // Weight A-fragments (4 k-steps), one 32B coalesced load each.
    v16h wfr[4];
#pragma unroll
    for (int kt = 0; kt < 4; ++kt)
      wfr[kt] = *(const v16h*)(afrag + (((size_t)kt * 64 + ntile) * 32 + lane) * 16);

    // Per-lane bias / conv2 weights for this lane's 8 consecutive n values.
    int nbase = ntile * 16 + hb8;
    float bias[8], w2l[24];
    v4f bv0 = *(const v4f*)&bc1_s[nbase];
    v4f bv1 = *(const v4f*)&bc1_s[nbase + 4];
#pragma unroll
    for (int i = 0; i < 4; ++i) { bias[i] = bv0[i]; bias[4 + i] = bv1[i]; }
#pragma unroll
    for (int i = 0; i < 6; ++i) {
      v4f wv = *(const v4f*)&w2_s[nbase * 3 + i * 4];
#pragma unroll
      for (int u = 0; u < 4; ++u) w2l[i * 4 + u] = wv[u];
    }

    for (int dtile = 0; dtile < 8; ++dtile) {
      v8f acc = {};
      int drow = dtile * 16 + nlocal;                 // B-frag: N(d) = lane%16
#pragma unroll
      for (int kt = 0; kt < 4; ++kt) {
        const _Float16* ab = &act_lds[drow * AST + kt * 32 + koffB];
        v8h lo = *(const v8h*)ab;                     // K = koffB + 0..7
        v8h hi = *(const v8h*)(ab + 16);              // K = 16 + koffB + 0..7
        v16h bf = __builtin_shufflevector(lo, hi, 0, 1, 2, 3, 4, 5, 6, 7,
                                          8, 9, 10, 11, 12, 13, 14, 15);
        acc = __builtin_amdgcn_wmma_f32_16x16x32_f16(
            false, wfr[kt], false, bf, (short)0, acc, false, false);
      }
      // Epilogue: bias+relu per n (VGPR dim), reduce conv2 over n in regs.
      float gM = 0.f, g0 = 0.f, gP = 0.f;             // -> logits[d-1], [d], [d+1]
#pragma unroll
      for (int r = 0; r < 8; ++r) {
        float h = fmaxf(acc[r] + bias[r], 0.f);
        gP += h * w2l[r * 3 + 0];                     // Wc2[n,0] -> logits[d+1]
        g0 += h * w2l[r * 3 + 1];                     // Wc2[n,1] -> logits[d]
        gM += h * w2l[r * 3 + 2];                     // Wc2[n,2] -> logits[d-1]
      }
      int d = dtile * 16 + nlocal;
      atomicAdd(&logit_s[half][d],     gM);           // logits index d' stored at d'+1
      atomicAdd(&logit_s[half][d + 1], g0);
      atomicAdd(&logit_s[half][d + 2], gP);
    }
  }
  __syncthreads();

  // ---- softmax + argmax-window expectation (threads 0..127 active) ----
  float bc2v = bc2[0];
  float l = 0.f;
  if (t < 128) {
    l = logit_s[0][t + 1] + logit_s[1][t + 1] + bc2v;
    red_v[t] = l;
    red_i[t] = t;
  }
  __syncthreads();
  for (int s = 64; s > 0; s >>= 1) {
    if (t < s) {
      float ov = red_v[t + s]; int oi = red_i[t + s];
      if (ov > red_v[t] || (ov == red_v[t] && oi < red_i[t])) {
        red_v[t] = ov; red_i[t] = oi;
      }
    }
    __syncthreads();
  }
  float maxv = red_v[0];
  int idx = red_i[0];
  __syncthreads();

  float ex = 0.f;
  if (t < 128) { ex = __expf(l - maxv); red_v[t] = ex; }
  __syncthreads();
  for (int s = 64; s > 0; s >>= 1) {
    if (t < s) red_v[t] += red_v[t + s];
    __syncthreads();
  }
  float sumall = red_v[0];
  __syncthreads();

  float vol = ex / sumall;
  int dd = t - idx; if (dd < 0) dd = -dd;
  float masked = (t < 128 && dd <= 5) ? vol : 0.f;    // KS//2 = 5
  if (t < 128) red_v[t] = masked;
  __syncthreads();
  for (int s = 64; s > 0; s >>= 1) {
    if (t < s) red_v[t] += red_v[t + s];
    __syncthreads();
  }
  float s2 = red_v[0];
  __syncthreads();

  if (t < 128) red_v[t] = masked / (s2 + 1e-8f) * (float)t;
  __syncthreads();
  for (int s = 64; s > 0; s >>= 1) {
    if (t < s) red_v[t] += red_v[t + s];
    __syncthreads();
  }
  if (t == 0) out[m] = red_v[0] * (6000.f / 127.f) + 2000.f;
}

// ---------------------------------------------------------------------------
extern "C" void kernel_launch(void* const* d_in, const int* in_sizes, int n_in,
                              void* d_out, int out_size, void* d_ws, size_t ws_size,
                              hipStream_t stream) {
  const float* kpts = (const float*)d_in[0];
  const float* jvis = (const float*)d_in[2];
  const float* camR = (const float*)d_in[5];
  const float* camT = (const float*)d_in[6];
  const float* camF = (const float*)d_in[7];
  const float* camC = (const float*)d_in[8];
  const float* imwh = (const float*)d_in[9];
  const int*   nper = (const int*)d_in[10];
  const float* Wr1  = (const float*)d_in[11];
  const float* br1  = (const float*)d_in[12];
  const float* Wr2  = (const float*)d_in[13];
  const float* br2  = (const float*)d_in[14];
  const float* Wc1  = (const float*)d_in[15];
  const float* bc1  = (const float*)d_in[16];
  const float* Wc2  = (const float*)d_in[17];
  const float* bc2  = (const float*)d_in[18];
  float* out = (float*)d_out;

  char* ws = (char*)d_ws;
  float*    rough_ws = (float*)ws;                              // 640*17*4
  float*    xin      = (float*)(ws + 65536);                    // 640*33*128*4
  _Float16* afrag    = (_Float16*)(ws + 65536 + (size_t)640 * KIN * ND_ * 4);

  prep_afrag<<<512, 256, 0, stream>>>(Wc1, afrag);
  rough_kernel<<<B_ * NP_, 256, 0, stream>>>(kpts, imwh, Wr1, br1, Wr2, br2, rough_ws);
  score_kernel<<<B_ * NP_, 128, 0, stream>>>(kpts, jvis, rough_ws, camR, camT,
                                             camF, camC, imwh, nper, xin);
  conv_head<<<B_ * NP_, 256, 0, stream>>>(xin, afrag, bc1, Wc2, bc2, out);
}
